// Hybrid_31602369364218
// MI455X (gfx1250) — compile-verified, matrix operand order checked by reading, not run on hardware
//
#include <hip/hip_runtime.h>
#include <hip/hip_fp16.h>
#include <math.h>

typedef __attribute__((ext_vector_type(16))) _Float16 v16h;
typedef __attribute__((ext_vector_type(8)))  float    v8f;

#define LN_EPS 1e-5f

// ======================= generic WMMA GEMM (16x64 per wave) =======================
// C = act(A[M,K] @ B[K,N] + bias).  A,B f32 row-major, f16 convert in-register,
// f32 accum via v_wmma_f32_16x16x32_f16.  N is compile-time so the 16 strided
// B loads become immediate-offset loads off one base pointer.  Each wave owns a
// 16(M) x 64(N) tile: A fragment reused across 4 WMMAs per k-step.
// If gridDim.z > 1: split-K; z-slice writes raw partials to C + z*M*N
// (bias/act skipped) -> deterministic (no float atomics).
template<int N, bool NG>
__global__ void gemm_wmma_kernel(const float* __restrict__ A,
                                 const float* __restrict__ B,
                                 const float* __restrict__ bias,
                                 float* __restrict__ C,
                                 int M, int K, int kchunk, int act)
{
  int lane  = threadIdx.x & 31;
  int row   = blockIdx.x * 16 + (lane & 15);
  int koff  = (lane >> 4) * 16;
  int nbase = blockIdx.y * 64;
  int k0 = blockIdx.z * kchunk;
  int k1 = k0 + kchunk; if (k1 > K) k1 = K;
  v8f acc[4] = {};
  for (int k = k0; k < k1; k += 32) {
    v16h a;
    const float* pa = A + (size_t)row * K + (k + koff);
    #pragma unroll
    for (int i = 0; i < 16; ++i) a[i] = (_Float16)pa[i];
    const float* pb0 = B + (size_t)(k + koff) * N + nbase + (lane & 15);
    #pragma unroll
    for (int s = 0; s < 4; ++s) {
      v16h b;
      if (!NG || (nbase + s * 16 + (lane & 15)) < N) {
        const float* pb = pb0 + s * 16;
        #pragma unroll
        for (int i = 0; i < 16; ++i) b[i] = (_Float16)pb[i * N];  // immediate offsets
      } else {
        #pragma unroll
        for (int i = 0; i < 16; ++i) b[i] = (_Float16)0.0f;
      }
      acc[s] = __builtin_amdgcn_wmma_f32_16x16x32_f16(false, a, false, b,
                                                      (short)0, acc[s], false, false);
    }
  }
  int mbase = blockIdx.x * 16 + (lane >> 4) * 8;
  float* Cz = C + (size_t)blockIdx.z * ((size_t)M * N);
  bool raw = (gridDim.z > 1);
  #pragma unroll
  for (int s = 0; s < 4; ++s) {
    int n = nbase + s * 16 + (lane & 15);
    if (NG && n >= N) continue;
    #pragma unroll
    for (int r = 0; r < 8; ++r) {
      int m = mbase + r;
      float v = acc[s][r];
      if (!raw) {
        if (bias) v += bias[n];
        if (act == 1) v = 0.5f * v * (1.0f + erff(v * 0.70710678118654752f)); // exact GELU
      }
      Cz[(size_t)m * N + n] = v;
    }
  }
}

template<int N, bool NG>
static void launch_gemm(const float* A, const float* B, const float* bias, float* C,
                        int M, int K, int act, int nsplit, hipStream_t stream)
{
  gemm_wmma_kernel<N, NG><<<dim3(M / 16, (N + 63) / 64, nsplit), dim3(32), 0, stream>>>(
      A, B, bias, C, M, K, K / nsplit, act);
}

// ======================= implicit-GEMM conv (WMMA) =======================
// out[b][o][l] = act(scale[o]*conv + shift[o]); conv over Cin*13 taps,
// 'same' pad 6, optional 2x nearest upsample folded into the gather.
// M = out channels, N = positions, K = Cin*13 (zero-filled to mult of 32).
__global__ void conv_wmma_kernel(const float* __restrict__ x,
                                 const float* __restrict__ w,
                                 const float* __restrict__ scale,
                                 const float* __restrict__ shift,
                                 float* __restrict__ out,
                                 int Cin, int Lin, int O, int rep, int act)
{
  int lane = threadIdx.x & 31;
  int bidx = blockIdx.z;
  int Leff = Lin * rep;
  int K = Cin * 13;
  int l    = blockIdx.x * 16 + (lane & 15);
  int mrow = blockIdx.y * 16 + (lane & 15);
  int koff = (lane >> 4) * 16;
  v8f acc = {};
  for (int k0 = 0; k0 < K; k0 += 32) {
    v16h a, bf;
    #pragma unroll
    for (int i = 0; i < 16; ++i) {
      int kk = k0 + koff + i;
      float av = 0.0f, bv = 0.0f;
      if (kk < K) {
        if (mrow < O) av = w[(size_t)mrow * K + kk];
        int c = kk / 13;
        int t = kk - c * 13;
        int src = l + t - 6;
        if (src >= 0 && src < Leff)
          bv = x[((size_t)(bidx * Cin + c)) * Lin + src / rep];
      }
      a[i]  = (_Float16)av;
      bf[i] = (_Float16)bv;
    }
    acc = __builtin_amdgcn_wmma_f32_16x16x32_f16(false, a, false, bf,
                                                 (short)0, acc, false, false);
  }
  int mbase = blockIdx.y * 16 + (lane >> 4) * 8;
  int n = blockIdx.x * 16 + (lane & 15);
  #pragma unroll
  for (int r = 0; r < 8; ++r) {
    int m = mbase + r;
    if (m >= O) continue;
    float v = acc[r] * scale[m] + shift[m];
    if (act == 1) v = fmaxf(v, 0.0f);
    else if (act == 2) v = (v > 0.0f) ? v : 0.01f * v;
    out[((size_t)(bidx * O + m)) * Leff + n] = v;
  }
}

// ======================= small elementwise / reduction kernels =======================
__global__ void prep_bn_kernel(const float* g, const float* b, const float* cb,
                               float* scale, float* shift, int n)
{
  int i = blockIdx.x * blockDim.x + threadIdx.x;
  if (i >= n) return;
  float s = g[i] * rsqrtf(1.0f + LN_EPS);
  scale[i] = s;
  shift[i] = b[i] + (cb ? cb[i] * s : 0.0f);
}

__global__ void maxpool2_kernel(const float* __restrict__ in, float* __restrict__ out,
                                int rows, int Lout)
{
  int i = blockIdx.x * blockDim.x + threadIdx.x;
  if (i >= rows * Lout) return;
  int r = i / Lout, l = i - r * Lout;
  const float* p = in + (size_t)r * (Lout * 2) + l * 2;
  out[i] = fmaxf(p[0], p[1]);
}

__global__ void dual_transpose_kernel(const float* __restrict__ z,
                                      float* __restrict__ x1, float* __restrict__ x2)
{
  // z: [64,256,T=256] -> x1=x2 : [64,T,256]
  int i = blockIdx.x * blockDim.x + threadIdx.x;
  if (i >= 64 * 256 * 256) return;
  int b = i >> 16;
  int r = i & 65535;
  int t = r >> 8, c = r & 255;
  float v = z[((size_t)(b * 256 + c)) * 256 + t];
  x1[i] = v; x2[i] = v;
}

__global__ void add_inplace_kernel(float* __restrict__ a, const float* __restrict__ b, int n)
{
  int i = blockIdx.x * blockDim.x + threadIdx.x;
  if (i < n) a[i] += b[i];
}

__global__ void layernorm_kernel(const float* __restrict__ x, const float* __restrict__ g,
                                 const float* __restrict__ b, float* __restrict__ out, int rows)
{
  int wid = threadIdx.x >> 5, lane = threadIdx.x & 31;
  int row = blockIdx.x * (blockDim.x >> 5) + wid;
  if (row >= rows) return;
  const float* xr = x + (size_t)row * 256;
  float v[8], s = 0.0f, ss = 0.0f;
  #pragma unroll
  for (int i = 0; i < 8; ++i) { v[i] = xr[lane + 32 * i]; s += v[i]; ss += v[i] * v[i]; }
  #pragma unroll
  for (int off = 16; off; off >>= 1) {
    s  += __shfl_xor(s,  off, 32);
    ss += __shfl_xor(ss, off, 32);
  }
  float m   = s * (1.0f / 256.0f);
  float var = ss * (1.0f / 256.0f) - m * m;
  float inv = rsqrtf(var + LN_EPS);
  float* orow = out + (size_t)row * 256;
  #pragma unroll
  for (int i = 0; i < 8; ++i) {
    int c = lane + 32 * i;
    orow[c] = (v[i] - m) * inv * g[c] + b[c];
  }
}

// ======================= LSH attention kernels =======================
// block = (bh, hash), 256 threads = t.  buckets[bh][h][t]
__global__ void lsh_hash_kernel(const float* __restrict__ qk, const float* __restrict__ rot,
                                int* __restrict__ buckets)
{
  int blk = blockIdx.x;
  int bh = blk >> 3, h = blk & 7;
  int t = threadIdx.x;
  int b = bh >> 3, hd = bh & 7;
  __shared__ float lrot[1024];                 // [d][r] slab for this hash
  for (int i = threadIdx.x; i < 1024; i += 256) {
    int d = i >> 5, r = i & 31;
    lrot[i] = rot[d * 256 + h * 32 + r];
  }
  __syncthreads();
  const float* q = qk + ((size_t)(b * 256 + t)) * 256 + hd * 32;
  float qv[32], s[32];
  #pragma unroll
  for (int d = 0; d < 32; ++d) qv[d] = q[d];
  #pragma unroll
  for (int r = 0; r < 32; ++r) s[r] = 0.0f;
  for (int d = 0; d < 32; ++d) {
    float qd = qv[d];
    #pragma unroll
    for (int r = 0; r < 32; ++r) s[r] += qd * lrot[d * 32 + r];
  }
  float best = -3.4e38f; int bi = 0;
  #pragma unroll
  for (int r = 0; r < 32; ++r) if (s[r] > best)  { best = s[r];  bi = r; }
  #pragma unroll
  for (int r = 0; r < 32; ++r) if (-s[r] > best) { best = -s[r]; bi = r + 32; }
  buckets[(size_t)bh * 2048 + h * 256 + t] = bi + h * 64;
}

// stable rank sort of 256 items per (bh, hash)
__global__ void lsh_sort_kernel(const int* __restrict__ buckets,
                                int* __restrict__ sortedT, int* __restrict__ undo)
{
  int blk = blockIdx.x;
  int bh = blk >> 3, h = blk & 7;
  int t = threadIdx.x;
  __shared__ int sb[256];
  int bv = buckets[(size_t)bh * 2048 + h * 256 + t];
  sb[t] = bv;
  __syncthreads();
  int pos = 0;
  for (int i = 0; i < 256; ++i) {
    int o = sb[i];
    pos += (o < bv) || (o == bv && i < t);
  }
  sortedT[(size_t)bh * 2048 + h * 256 + pos] = t;
  undo[(size_t)bh * 2048 + h * 256 + t] = h * 256 + pos;
}

// one thread per sorted query position; 4-wide chunks, look-one-back over 512-ring
__global__ void lsh_attn_kernel(const float* __restrict__ qk, const float* __restrict__ v,
                                const int* __restrict__ sortedT,
                                float* __restrict__ so, float* __restrict__ slog)
{
  int gid = blockIdx.x * blockDim.x + threadIdx.x;   // 512*2048
  int bh = gid >> 11, j = gid & 2047;
  int b = bh >> 3, hd = bh & 7;
  const int* st = sortedT + (size_t)bh * 2048;
  int tq = st[j];
  const float* base = qk + (size_t)(b * 256) * 256 + hd * 32;
  float q[32];
  const float* qr = base + (size_t)tq * 256;
  #pragma unroll
  for (int d = 0; d < 32; ++d) q[d] = qr[d];
  int chunk = j >> 2;
  int prev  = (chunk + 511) & 511;
  float dots[8]; int tks[8];
  #pragma unroll
  for (int jj = 0; jj < 8; ++jj) {
    int ck = (jj < 4) ? chunk : prev;
    int tk = st[ck * 4 + (jj & 3)];
    tks[jj] = tk;
    const float* kr = base + (size_t)tk * 256;
    float nrm = 0.0f, dt = 0.0f;
    for (int d = 0; d < 32; ++d) { float kv = kr[d]; nrm += kv * kv; dt += q[d] * kv; }
    nrm = fmaxf(sqrtf(nrm), 1e-12f);
    dt = dt / nrm * 0.1767766952966369f;     // 32^-0.5
    if (tq < tk)       dt = -1e9f;           // causal
    else if (tq == tk) dt = -5e4f;           // self mask
    dots[jj] = dt;
  }
  float m = dots[0];
  #pragma unroll
  for (int jj = 1; jj < 8; ++jj) m = fmaxf(m, dots[jj]);
  float sum = 0.0f;
  #pragma unroll
  for (int jj = 0; jj < 8; ++jj) sum += expf(dots[jj] - m);
  float lse = m + logf(sum);
  float o[32];
  #pragma unroll
  for (int d = 0; d < 32; ++d) o[d] = 0.0f;
  const float* vbase = v + (size_t)(b * 256) * 256 + hd * 32;
  #pragma unroll
  for (int jj = 0; jj < 8; ++jj) {
    float p = expf(dots[jj] - lse);
    const float* vr = vbase + (size_t)tks[jj] * 256;
    for (int d = 0; d < 32; ++d) o[d] += p * vr[d];
  }
  float* orow = so + (size_t)gid * 32;
  for (int d = 0; d < 32; ++d) orow[d] = o[d];
  slog[gid] = lse;
}

// unsort + softmax-weighted combine over 8 hash rounds
__global__ void lsh_combine_kernel(const float* __restrict__ so, const float* __restrict__ slog,
                                   const int* __restrict__ undo, float* __restrict__ ao)
{
  int gid = blockIdx.x * blockDim.x + threadIdx.x;  // 512*256
  int bh = gid >> 8, t = gid & 255;
  int b = bh >> 3, hd = bh & 7;
  float lg[8]; int jj[8];
  #pragma unroll
  for (int h = 0; h < 8; ++h) {
    int u = undo[(size_t)bh * 2048 + h * 256 + t];
    jj[h] = u;
    lg[h] = slog[(size_t)bh * 2048 + u];
  }
  float m = lg[0];
  #pragma unroll
  for (int h = 1; h < 8; ++h) m = fmaxf(m, lg[h]);
  float sum = 0.0f;
  #pragma unroll
  for (int h = 0; h < 8; ++h) sum += expf(lg[h] - m);
  float inv = 1.0f / sum;
  float acc[32];
  #pragma unroll
  for (int d = 0; d < 32; ++d) acc[d] = 0.0f;
  #pragma unroll
  for (int h = 0; h < 8; ++h) {
    float w = expf(lg[h] - m) * inv;
    const float* sr = so + ((size_t)bh * 2048 + jj[h]) * 32;
    for (int d = 0; d < 32; ++d) acc[d] += w * sr[d];
  }
  float* orow = ao + (size_t)(b * 256 + t) * 256 + hd * 32;
  for (int d = 0; d < 32; ++d) orow[d] = acc[d];
}

// zr = 0.5*(x1+x2), channel-major flatten: flat[b][c*256+t]
__global__ void flat_kernel(const float* __restrict__ x1, const float* __restrict__ x2,
                            float* __restrict__ flat)
{
  int gid = blockIdx.x * blockDim.x + threadIdx.x;
  if (gid >= 64 * 65536) return;
  int b = gid >> 16;
  int r = gid & 65535;
  int c = r >> 8, t = r & 255;
  size_t idx = ((size_t)(b * 256 + t)) * 256 + c;
  flat[gid] = 0.5f * (x1[idx] + x2[idx]);
}

// reduce split-K partials + classifier BN + ReLU
__global__ void reduce_relu_kernel(const float* __restrict__ part,
                                   const float* __restrict__ scale, const float* __restrict__ shift,
                                   float* __restrict__ out, int MN, int nsplit, int N)
{
  int i = blockIdx.x * blockDim.x + threadIdx.x;
  if (i >= MN) return;
  int n = i % N;
  float s = 0.0f;
  for (int k = 0; k < nsplit; ++k) s += part[(size_t)k * MN + i];
  out[i] = fmaxf(s * scale[n] + shift[n], 0.0f);
}

// ======================= host launch sequence =======================
extern "C" void kernel_launch(void* const* d_in, const int* in_sizes, int n_in,
                              void* d_out, int out_size, void* d_ws, size_t ws_size,
                              hipStream_t stream)
{
  (void)in_sizes; (void)out_size; (void)ws_size;
  const float* P[60];
  for (int i = 0; i < 60 && i < n_in; ++i) P[i] = (const float*)d_in[i];
  float* out = (float*)d_out;

  float* W = (float*)d_ws;
  size_t off = 0;
  auto alloc = [&](size_t n) { float* p = W + off; off += n; return p; };
  float* buf1 = alloc(8388608);    // [64,128,1024] slot (e1,e2,d3,flat)
  float* buf2 = alloc(4194304);
  float* buf3 = alloc(4194304);
  float* buf4 = alloc(4194304);
  float* x1   = alloc(4194304);
  float* x2   = alloc(4194304);
  float* buf7 = alloc(4194304);
  float* ffh  = alloc(16777216);   // [16384,1024]
  float* so   = alloc(33554432);   // [512*2048,32]
  float* slog = alloc(1048576);
  float* part = alloc(4194304);    // 64 splits x [64,1024]
  float* c1   = alloc(65536);
  float* scl  = alloc(1024);
  float* shf  = alloc(1024);
  int* bucketsI = (int*)alloc(1048576);
  int* sortedT  = (int*)alloc(1048576);
  int* undoI    = (int*)alloc(1048576);

  auto convL = [&](const float* x, const float* w, const float* g, const float* bnb,
                   const float* cb, int Cin, int Lin, int O, int rep, int act, float* o) {
    prep_bn_kernel<<<dim3((O + 255) / 256), dim3(256), 0, stream>>>(g, bnb, cb, scl, shf, O);
    int Leff = Lin * rep;
    conv_wmma_kernel<<<dim3(Leff / 16, (O + 15) / 16, 64), dim3(32), 0, stream>>>(
        x, w, scl, shf, o, Cin, Lin, O, rep, act);
  };

  // ---- encoder ----
  convL(P[0],  P[1],  P[2],  P[3],  nullptr, 2,   1024, 128, 1, 1, buf1);
  maxpool2_kernel<<<dim3((64*128*512 + 255)/256), dim3(256), 0, stream>>>(buf1, buf2, 64*128, 512);
  convL(buf2,  P[4],  P[5],  P[6],  nullptr, 128, 512,  256, 1, 1, buf1);
  maxpool2_kernel<<<dim3((64*256*256 + 255)/256), dim3(256), 0, stream>>>(buf1, buf2, 64*256, 256);
  convL(buf2,  P[7],  P[8],  P[9],  nullptr, 256, 256,  256, 1, 1, buf3);
  convL(buf3,  P[10], P[11], P[12], nullptr, 256, 256,  256, 1, 1, buf4);   // z
  // ---- decoder ----
  convL(buf4,  P[13], P[14], P[15], nullptr, 256, 256,  128, 1, 2, buf2);   // d1
  convL(buf2,  P[16], P[18], P[19], P[17],   128, 256,  128, 2, 2, buf3);   // d2 (rep2)
  convL(buf3,  P[20], P[22], P[23], P[21],   128, 512,  64,  2, 2, buf1);   // d3 (rep2)
  convL(buf1,  P[24], P[26], P[27], P[25],   64,  1024, 2,   1, 0, out + 1536); // recon

  // ---- reformer ----
  dual_transpose_kernel<<<dim3(16384), dim3(256), 0, stream>>>(buf4, x1, x2);
  const int M = 16384;
  auto layer = [&](int ib, const float* rot) {
    layernorm_kernel<<<dim3(M / 8), dim3(256), 0, stream>>>(x2, P[ib+0], P[ib+1], buf4, M);
    launch_gemm<256, false>(buf4, P[ib+2], nullptr, buf7, M, 256, 0, 1, stream);  // qk
    launch_gemm<256, false>(buf4, P[ib+3], nullptr, buf2, M, 256, 0, 1, stream);  // v
    lsh_hash_kernel<<<dim3(4096), dim3(256), 0, stream>>>(buf7, rot, bucketsI);
    lsh_sort_kernel<<<dim3(4096), dim3(256), 0, stream>>>(bucketsI, sortedT, undoI);
    lsh_attn_kernel<<<dim3(4096), dim3(256), 0, stream>>>(buf7, buf2, sortedT, so, slog);
    lsh_combine_kernel<<<dim3(512), dim3(256), 0, stream>>>(so, slog, undoI, buf3);
    launch_gemm<256, false>(buf3, P[ib+4], P[ib+5], buf4, M, 256, 0, 1, stream);  // out proj
    add_inplace_kernel<<<dim3(16384), dim3(256), 0, stream>>>(x1, buf4, M * 256);
    layernorm_kernel<<<dim3(M / 8), dim3(256), 0, stream>>>(x1, P[ib+6], P[ib+7], buf4, M);
    launch_gemm<1024, false>(buf4, P[ib+8],  P[ib+9],  ffh,  M, 256,  1, 1, stream); // FF1+GELU
    launch_gemm<256,  false>(ffh,  P[ib+10], P[ib+11], buf7, M, 1024, 0, 1, stream); // FF2
    add_inplace_kernel<<<dim3(16384), dim3(256), 0, stream>>>(x2, buf7, M * 256);
  };
  layer(28, P[58]);
  layer(40, P[59]);

  // ---- classifier ----
  flat_kernel<<<dim3(16384), dim3(256), 0, stream>>>(x1, x2, buf1);
  launch_gemm<1024, false>(buf1, P[52], nullptr, part, 64, 65536, 0, 64, stream);  // split-K 64
  prep_bn_kernel<<<dim3(4), dim3(256), 0, stream>>>(P[54], P[55], P[53], scl, shf, 1024);
  reduce_relu_kernel<<<dim3(256), dim3(256), 0, stream>>>(part, scl, shf, c1, 64*1024, 64, 1024);
  launch_gemm<24, true>(c1, P[56], P[57], out, 64, 1024, 0, 1, stream);            // y
}
